// Stage1_65695819760184
// MI455X (gfx1250) — matrix operand order repaired and checked by reading, not compiled, not run on hardware
//
#include <hip/hip_runtime.h>
#include <hip/hip_bf16.h>

// ---------------------------------------------------------------------------
// Types for CDNA5 WMMA (wave32): D(16x16 f32) = A(16x32 bf16) x B(32x16 bf16) + C
// ---------------------------------------------------------------------------
typedef __bf16 bf16x8  __attribute__((ext_vector_type(8)));
typedef __bf16 v16bf   __attribute__((ext_vector_type(16)));
typedef float  v8f     __attribute__((ext_vector_type(8)));

// ---------------------------------------------------------------------------
// fp32 -> bf16 convert (RNE via clang __bf16 conversion)
// ---------------------------------------------------------------------------
__global__ void k_f32_to_bf16(const float* __restrict__ x, __bf16* __restrict__ y,
                              size_t n) {
    size_t i = (size_t)blockIdx.x * blockDim.x + threadIdx.x;
    if (i < n) y[i] = (__bf16)x[i];
}

// ---------------------------------------------------------------------------
// Weight transpose+convert: w[k][i][o] (fp32) -> wt[k][o][i] (bf16)
// so B-matrix columns (output channels) are contiguous along K.
// ---------------------------------------------------------------------------
__global__ void k_transpose_w(const float* __restrict__ w, __bf16* __restrict__ wt,
                              int Cin, int Cout, size_t total) {
    size_t id = (size_t)blockIdx.x * blockDim.x + threadIdx.x;
    if (id >= total) return;
    int    i    = (int)(id % Cin);
    size_t rest = id / Cin;
    int    o    = (int)(rest % Cout);
    int    k    = (int)(rest / Cout);
    wt[id] = (__bf16)w[((size_t)k * Cin + i) * Cout + o];
}

// ---------------------------------------------------------------------------
// Sparse conv gather-WMMA-scatter, 16x64 output tile per wave.
//   grid = (ceil(P/64), Cout/64, 27), block = 128 (4 waves)
//   wave w handles pair-row tile blockIdx.x*4+w and 64 output cols.
// Each K-step loads one A fragment (shared) + 4 B fragments -> 4 WMMAs,
// quartering gather traffic for the random A rows vs a 16x16 tile.
//
// A layout (16-bit 16x32): lane&15 = M row; lanes 0-15 hold K = K0+0..7 and
// K0+16..23; lanes 16-31 hold K = K0+8..15 and K0+24..31.
// B layout (16-bit 32x16): lane&15 = N col; lanes 0-15 hold K = K0+0..7 and
// K0+16..23; lanes 16-31 hold K = K0+8..15 and K0+24..31 (same as A).
// C/D layout: VGPR v -> row M = v + 8*(lane>=16), col N = lane&15.
// ---------------------------------------------------------------------------
template <int CIN>
__global__ void __launch_bounds__(128)
k_sparse_conv_wmma(const __bf16* __restrict__ x, const __bf16* __restrict__ wt,
                   const int* __restrict__ in_idx, const int* __restrict__ out_idx,
                   int P, int Cout, float* __restrict__ out) {
    const int k       = blockIdx.z;
    const int colBase = blockIdx.y * 64;
    const int wave    = threadIdx.x >> 5;
    const int lane    = threadIdx.x & 31;
    const int tileRow = blockIdx.x * 4 + wave;
    if (tileRow * 16 >= P) return;            // wave-uniform exit (EXEC all-1 for WMMA)

    const int half = lane >> 4;               // 0: lanes 0-15, 1: lanes 16-31
    const int nidx = lane & 15;

    const int* kin  = in_idx  + (size_t)k * P;
    const int* kout = out_idx + (size_t)k * P;

    const int pr   = tileRow * 16 + nidx;
    const int grow = (pr < P) ? kin[pr] : 0;  // padded lanes gather row 0 (discarded)
    const __bf16* xrow  = x  + (size_t)grow * CIN;
    const __bf16* wrow0 = wt + ((size_t)k * Cout + (size_t)colBase + nidx) * (size_t)CIN;

    v8f c[4] = {v8f{}, v8f{}, v8f{}, v8f{}};
#pragma unroll 2
    for (int K0 = 0; K0 < CIN; K0 += 32) {
        // A: two 16-byte loads per lane, shared by all 4 column tiles
        bf16x8 a0 = *(const bf16x8*)(xrow + K0 + half * 8);
        bf16x8 a1 = *(const bf16x8*)(xrow + K0 + 16 + half * 8);
        v16bf A = __builtin_shufflevector(a0, a1, 0,1,2,3,4,5,6,7,8,9,10,11,12,13,14,15);
#pragma unroll
        for (int j = 0; j < 4; ++j) {
            const __bf16* wrow = wrow0 + (size_t)j * 16 * CIN;
            bf16x8 b0 = *(const bf16x8*)(wrow + K0 + half * 8);
            bf16x8 b1 = *(const bf16x8*)(wrow + K0 + 16 + half * 8);
            v16bf B = __builtin_shufflevector(b0, b1, 0,1,2,3,4,5,6,7,8,9,10,11,12,13,14,15);
            c[j] = __builtin_amdgcn_wmma_f32_16x16x32_bf16(false, A, false, B,
                                                           (short)0, c[j], false, false);
        }
    }

#pragma unroll
    for (int v = 0; v < 8; ++v) {
        int m   = v + half * 8;
        int prr = tileRow * 16 + m;
        if (prr < P) {
            int orow = kout[prr];             // sentinel row n_out collects padding
            float* orp = out + (size_t)orow * Cout + colBase + nidx;
#pragma unroll
            for (int j = 0; j < 4; ++j)
                atomicAdd(orp + j * 16, c[j][v]);
        }
    }
}

// ---------------------------------------------------------------------------
// Per-column sum / sumsq partial reduction (C = 512 fixed). 64 rows per block,
// 256 threads -> each thread owns columns t and t+256 (coalesced row reads).
// ---------------------------------------------------------------------------
__global__ void k_col_stats_partial(const float* __restrict__ x, int nrows,
                                    float* __restrict__ sum, float* __restrict__ sq) {
    const int t  = threadIdx.x;
    const int r0 = blockIdx.x * 64;
    const int r1 = min(r0 + 64, nrows);
    float s0 = 0.f, q0 = 0.f, s1 = 0.f, q1 = 0.f;
    for (int r = r0; r < r1; ++r) {
        const float* row = x + (size_t)r * 512;
        float a = row[t];       s0 += a; q0 += a * a;
        float b = row[t + 256]; s1 += b; q1 += b * b;
    }
    atomicAdd(&sum[t], s0);       atomicAdd(&sq[t], q0);
    atomicAdd(&sum[t + 256], s1); atomicAdd(&sq[t + 256], q1);
}

__global__ void k_col_stats_final(const float* __restrict__ sum, const float* __restrict__ sq,
                                  int n, float* __restrict__ mu, float* __restrict__ rstd) {
    int c = blockIdx.x * blockDim.x + threadIdx.x;
    if (c < 512) {
        float m = sum[c] / (float)n;
        float v = sq[c] / (float)n - m * m;       // biased variance (matches x.var(0))
        mu[c]   = m;
        rstd[c] = rsqrtf(v + 1e-5f);
    }
}

// BN + ReLU, write bf16 activations for the next conv's A-gather.
__global__ void k_bn_relu_apply(const float* __restrict__ x, size_t n,
                                const float* __restrict__ mu, const float* __restrict__ rstd,
                                const float* __restrict__ g, const float* __restrict__ b,
                                __bf16* __restrict__ y) {
    size_t i = (size_t)blockIdx.x * blockDim.x + threadIdx.x;
    if (i >= n) return;
    int   c = (int)(i & 511);
    float v = (x[i] - mu[c]) * rstd[c] * g[c] + b[c];
    y[i] = (__bf16)(v > 0.f ? v : 0.f);
}

// ---------------------------------------------------------------------------
// Segment max. Values are post-ReLU (>= 0) so uint-bit atomicMax on an
// fp32 buffer zero-initialized to 0 is exact.
// ---------------------------------------------------------------------------
__device__ __forceinline__ void atomic_max_pos(float* addr, float v) {
    atomicMax((unsigned int*)addr, __float_as_uint(v));
}

__global__ void k_seg_max_bf16(const __bf16* __restrict__ x, const int* __restrict__ idx,
                               int nIn, float* __restrict__ out) {
    size_t t = (size_t)blockIdx.x * blockDim.x + threadIdx.x;
    if (t >= (size_t)nIn * 512) return;
    int r = (int)(t >> 9), c = (int)(t & 511);
    atomic_max_pos(&out[(size_t)idx[r] * 512 + c], (float)x[t]);
}

__global__ void k_seg_max_f32(const float* __restrict__ x, const int* __restrict__ idx,
                              int nIn, float* __restrict__ out) {
    size_t t = (size_t)blockIdx.x * blockDim.x + threadIdx.x;
    if (t >= (size_t)nIn * 512) return;
    int r = (int)(t >> 9), c = (int)(t & 511);
    atomic_max_pos(&out[(size_t)idx[r] * 512 + c], x[t]);
}

// ---------------------------------------------------------------------------
// Dense FC (tiny: M = 8 rows). One thread per output element.
// ---------------------------------------------------------------------------
__global__ void k_fc(const float* __restrict__ x, const float* __restrict__ w,
                     const float* __restrict__ b, float* __restrict__ y,
                     int M, int K, int N, int relu) {
    int t = blockIdx.x * blockDim.x + threadIdx.x;
    if (t >= M * N) return;
    int m = t / N, n = t % N;
    float acc = b[n];
    const float* xr = x + (size_t)m * K;
    for (int k = 0; k < K; ++k) acc += xr[k] * w[(size_t)k * N + n];
    if (relu && acc < 0.f) acc = 0.f;
    y[t] = acc;
}

// ---------------------------------------------------------------------------
// Host orchestration (all on `stream`; graph-capture safe).
// ---------------------------------------------------------------------------
extern "C" void kernel_launch(void* const* d_in, const int* in_sizes, int n_in,
                              void* d_out, int out_size, void* d_ws, size_t ws_size,
                              hipStream_t stream) {
    const float* feats = (const float*)d_in[0];
    const float* w[6]  = {(const float*)d_in[1], (const float*)d_in[2], (const float*)d_in[3],
                          (const float*)d_in[4], (const float*)d_in[5], (const float*)d_in[6]};
    const float* bn_g  = (const float*)d_in[7];
    const float* bn_b  = (const float*)d_in[8];
    const float* fc1w  = (const float*)d_in[9];
    const float* fc1b  = (const float*)d_in[10];
    const float* fc2w  = (const float*)d_in[11];
    const float* fc2b  = (const float*)d_in[12];
    const float* fc3w  = (const float*)d_in[13];
    const float* fc3b  = (const float*)d_in[14];
    const int* m1in  = (const int*)d_in[15];
    const int* m1out = (const int*)d_in[16];
    const int* m2in  = (const int*)d_in[17];
    const int* m2out = (const int*)d_in[18];
    const int* p1idx = (const int*)d_in[19];
    const int* p2idx = (const int*)d_in[20];
    const int* bidx  = (const int*)d_in[21];

    const int n1 = in_sizes[0] / 256;        // level-1 active sites
    const int P1 = in_sizes[15] / 27;        // padded pairs/offset, level 1
    const int P2 = in_sizes[17] / 27;        // padded pairs/offset, level 2
    const int n2 = in_sizes[20];             // len(pool2_idx) == n2
    const int n3 = in_sizes[21];             // len(batch_idx) == n3

    // ---- workspace carve-up (256B aligned) ----
    char* wsp = (char*)d_ws;
    size_t off = 0;
    auto carve = [&](size_t bytes) -> void* {
        void* p = wsp + off;
        off = (off + bytes + 255) & ~(size_t)255;
        return p;
    };
    __bf16* act   = (__bf16*)carve((size_t)(n1 + 1) * 512 * sizeof(__bf16));
    float*  acc   = (float*) carve((size_t)(n1 + 1) * 512 * sizeof(float));
    __bf16* wt    = (__bf16*)carve((size_t)27 * 512 * 512 * sizeof(__bf16));
    float*  csum  = (float*) carve(512 * sizeof(float));
    float*  csq   = (float*) carve(512 * sizeof(float));
    float*  mu    = (float*) carve(512 * sizeof(float));
    float*  rstd  = (float*) carve(512 * sizeof(float));
    float*  bmax  = (float*) carve(8 * 512 * sizeof(float));
    float*  fcb1  = (float*) carve(8 * 4096 * sizeof(float));
    float*  fcb2  = (float*) carve(8 * 4096 * sizeof(float));
    (void)ws_size; (void)n_in; (void)out_size;

    auto run_conv = [&](const float* wsrc, int Cin, const int* min_, const int* mout_,
                        int P, int nrows, int layer) {
        size_t tw = (size_t)27 * Cin * 512;
        k_transpose_w<<<(unsigned)((tw + 255) / 256), 256, 0, stream>>>(wsrc, wt, Cin, 512, tw);
        hipMemsetAsync(acc, 0, (size_t)(nrows + 1) * 512 * sizeof(float), stream);
        dim3 g((P + 63) / 64, 512 / 64, 27);
        if (Cin == 256)
            k_sparse_conv_wmma<256><<<g, 128, 0, stream>>>(act, wt, min_, mout_, P, 512, acc);
        else
            k_sparse_conv_wmma<512><<<g, 128, 0, stream>>>(act, wt, min_, mout_, P, 512, acc);
        hipMemsetAsync(csum, 0, 512 * sizeof(float), stream);
        hipMemsetAsync(csq,  0, 512 * sizeof(float), stream);
        k_col_stats_partial<<<(nrows + 63) / 64, 256, 0, stream>>>(acc, nrows, csum, csq);
        k_col_stats_final<<<2, 256, 0, stream>>>(csum, csq, nrows, mu, rstd);
        size_t ne = (size_t)nrows * 512;
        k_bn_relu_apply<<<(unsigned)((ne + 255) / 256), 256, 0, stream>>>(
            acc, ne, mu, rstd, bn_g + (size_t)layer * 512, bn_b + (size_t)layer * 512, act);
    };

    // feats (fp32, n1 x 256) -> bf16 activations
    {
        size_t nf = (size_t)n1 * 256;
        k_f32_to_bf16<<<(unsigned)((nf + 255) / 256), 256, 0, stream>>>(feats, act, nf);
    }

    // level 1: three convs
    run_conv(w[0], 256, m1in, m1out, P1, n1, 0);
    run_conv(w[1], 512, m1in, m1out, P1, n1, 1);
    run_conv(w[2], 512, m1in, m1out, P1, n1, 2);

    // pool1: n1 -> n2
    hipMemsetAsync(acc, 0, (size_t)n2 * 512 * sizeof(float), stream);
    {
        size_t ne = (size_t)n1 * 512;
        k_seg_max_bf16<<<(unsigned)((ne + 255) / 256), 256, 0, stream>>>(act, p1idx, n1, acc);
        size_t np = (size_t)n2 * 512;
        k_f32_to_bf16<<<(unsigned)((np + 255) / 256), 256, 0, stream>>>(acc, act, np);
    }

    // level 2: three convs
    run_conv(w[3], 512, m2in, m2out, P2, n2, 3);
    run_conv(w[4], 512, m2in, m2out, P2, n2, 4);
    run_conv(w[5], 512, m2in, m2out, P2, n2, 5);

    // pool2: n2 -> n3 (keep fp32 in acc)
    hipMemsetAsync(acc, 0, (size_t)n3 * 512 * sizeof(float), stream);
    {
        size_t ne = (size_t)n2 * 512;
        k_seg_max_bf16<<<(unsigned)((ne + 255) / 256), 256, 0, stream>>>(act, p2idx, n2, acc);
    }

    // global batch max: n3 -> 8
    hipMemsetAsync(bmax, 0, 8 * 512 * sizeof(float), stream);
    {
        size_t ne = (size_t)n3 * 512;
        k_seg_max_f32<<<(unsigned)((ne + 255) / 256), 256, 0, stream>>>(acc, bidx, n3, bmax);
    }

    // FC head (fp32, M = 8)
    k_fc<<<(8 * 4096 + 255) / 256, 256, 0, stream>>>(bmax, fc1w, fc1b, fcb1, 8, 512, 4096, 1);
    k_fc<<<(8 * 4096 + 255) / 256, 256, 0, stream>>>(fcb1, fc2w, fc2b, fcb2, 8, 4096, 4096, 1);
    k_fc<<<(8 * 40 + 255) / 256, 256, 0, stream>>>(fcb2, fc3w, fc3b, (float*)d_out, 8, 4096, 40, 0);
}
